// TimeSformerBlock_56659208569095
// MI455X (gfx1250) — matrix operand order repaired and checked by reading, not compiled, not run on hardware
//
#include <hip/hip_runtime.h>
#include <hip/hip_bf16.h>
#include <hip/hip_fp16.h>

// ---------------------------------------------------------------------------
// CDNA5 (gfx1250) WMMA types
// ---------------------------------------------------------------------------
typedef __attribute__((ext_vector_type(16))) _Float16 v16h;
typedef __attribute__((ext_vector_type(8)))  float    v8f;
typedef __attribute__((ext_vector_type(8)))  _Float16 h8v;   // 16-byte f16 chunk

union HFrag { v16h v; _Float16 h[16]; };

#ifndef __has_builtin
#define __has_builtin(x) 0
#endif
#if __has_builtin(__builtin_amdgcn_tensor_load_to_lds)
#define HAS_TDM 1
#else
#define HAS_TDM 0
#endif

#if HAS_TDM
typedef unsigned int u32x4 __attribute__((ext_vector_type(4)));
typedef int          i32x4 __attribute__((ext_vector_type(4)));
typedef int          i32x8 __attribute__((ext_vector_type(8)));

// TDM descriptor (ISA 8.3/8.4): 2-D tile of f16, tile_dim0=32 (64B rows),
// tile_dim1=128 rows, tensor stride K halves, LDS padding 4 DWORDs after
// every 16 DWORDs -> padded LDS row stride of 40 halves (matches LDH).
__device__ __forceinline__ void tdm_load_tile_a(const _Float16* gsrc,
                                                unsigned lds_off, int K) {
  unsigned long long ga = (unsigned long long)(size_t)gsrc;
  u32x4 g0;
  g0[0] = 1u;                                          // count=1 (user D#)
  g0[1] = lds_off;                                     // lds_addr
  g0[2] = (unsigned)(ga & 0xffffffffull);              // global_addr[31:0]
  g0[3] = (unsigned)((ga >> 32) & 0x01ffffffull) | (2u << 30); // addr[56:32], type=2
  i32x8 g1;
  g1[0] = (1 << 16) | (1 << 20) | (3 << 22) | (3 << 25); // 2B, pad_en, 16DW/4DW
  g1[1] = (int)(((unsigned)K & 0xffffu) << 16);          // tensor_dim0 lo
  g1[2] = (int)((((unsigned)K >> 16) & 0xffffu) | (128u << 16)); // dim0 hi | dim1 lo
  g1[3] = (int)(32u << 16);                              // dim1 hi=0 | tile_dim0=32
  g1[4] = 128;                                           // tile_dim1=128, tile_dim2=0
  g1[5] = K;                                             // dim0_stride[31:0]
  g1[6] = 0;                                             // stride hi / dim1_stride lo
  g1[7] = 0;
  i32x4 gz = {0, 0, 0, 0};
#if __has_include(<hip/amd_detail/amd_gfx1250_TDM.h>)
  i32x8 gz8 = {0, 0, 0, 0, 0, 0, 0, 0};
  __builtin_amdgcn_tensor_load_to_lds(g0, g1, gz, gz, gz8, 0);
#else
  __builtin_amdgcn_tensor_load_to_lds(g0, g1, gz, gz, 0);
#endif
}
#endif  // HAS_TDM

// A-matrix fragment (16x32 f16):
//   lanes 0-15 : row M = lane,    K = 0..15
//   lanes16-31 : row M = lane-16, K = 16..31
__device__ __forceinline__ v16h load_a_frag(const _Float16* base, int stride,
                                            int mbase, int ks, int lane) {
  const int half = lane >> 4, lr = lane & 15;
  const _Float16* r = base + (size_t)(mbase + lr) * stride;
  HFrag f;
#pragma unroll
  for (int i = 0; i < 8; ++i) f.h[i]     = r[ks + half * 16 + i];
#pragma unroll
  for (int i = 0; i < 8; ++i) f.h[8 + i] = r[ks + half * 16 + 8 + i];
  return f.v;
}

// B-matrix fragment (32x16 f16): lane holds column N = nbase + lane%16;
// lanes 0-15 K=0..15, lanes 16-31 K=16..31. baseT is N-major [n][k].
__device__ __forceinline__ v16h load_b_frag(const _Float16* baseT, int stride,
                                            int nbase, int ks, int lane) {
  const int half = lane >> 4, lr = lane & 15;
  const _Float16* r = baseT + (size_t)(nbase + lr) * stride + ks + half * 16;
  HFrag f;
#pragma unroll
  for (int i = 0; i < 16; ++i) f.h[i] = r[i];
  return f.v;
}

__device__ __forceinline__ float gelu_exact(float v) {
  return 0.5f * v * (1.0f + erff(v * 0.70710678118654752f));
}

// ---------------------------------------------------------------------------
// f32 -> f16 conversion pass (one-shot for x and all weight matrices)
// ---------------------------------------------------------------------------
__global__ __launch_bounds__(256) void cvt_f32_f16(const float* __restrict__ s,
                                                   _Float16* __restrict__ d, int n) {
  const int i = (blockIdx.x * 256 + threadIdx.x) * 4;
  if (i < n) {
    const float4 q = *(const float4*)(s + i);
    d[i + 0] = (_Float16)q.x; d[i + 1] = (_Float16)q.y;
    d[i + 2] = (_Float16)q.z; d[i + 3] = (_Float16)q.w;
  }
}

// ---------------------------------------------------------------------------
// GEMM: C[M,N] = act(A[M,K](f16) @ W[K,N](f16) + bias[N] (+ resid[M,N] f32))
// 128x128x32 block tile, 8 waves (2x4), wave tile 64x32, double-buffered LDS.
// A tile fed by the Tensor Data Mover (async, TENSORcnt) when available.
// Output to f32 (Cf) or f16 (Ch), exactly one non-null.
// ---------------------------------------------------------------------------
__global__ __launch_bounds__(256) void gemm_bias_act(
    const _Float16* __restrict__ A, const _Float16* __restrict__ W,
    const float* __restrict__ bias, const float* __restrict__ resid,
    float* __restrict__ Cf, _Float16* __restrict__ Ch,
    int M, int N, int K, int act) {
  constexpr int BK = 32;
  constexpr int LDH = 40;                    // padded row stride (halves)
  __shared__ _Float16 sA[2][128 * LDH];      // [m][k]
  __shared__ _Float16 sB[2][128 * LDH];      // [n][k] (transposed store)

  const int tid  = threadIdx.x;
  const int wave = tid >> 5, lane = tid & 31;
  const int half = lane >> 4, lr = lane & 15;
  const int wm = wave >> 2, wn = wave & 3;   // 2x4 wave grid
  const int m0 = blockIdx.y * 128, n0 = blockIdx.x * 128;
  const int wave_m = wm * 64, wave_n = wn * 32;

  v8f acc[4][2];
#pragma unroll
  for (int i = 0; i < 4; ++i)
#pragma unroll
    for (int j = 0; j < 2; ++j)
#pragma unroll
      for (int e = 0; e < 8; ++e) acc[i][j][e] = 0.0f;

  // ---- stage helpers: B tile 32x128 f16 = 512 chunks of 8 halves ----
  // chunk c: kr = c>>4 (k row), nc = (c&15)*8 (col). 2 chunks per thread.
  h8v rb[2];
#if !HAS_TDM
  h8v ra[2];  // A tile fallback: chunk c: row = c>>2, kc = (c&3)*8
#endif

  // ---- prologue: tile 0 ----
#if HAS_TDM
  if (wave == 0)
    tdm_load_tile_a(A + (size_t)m0 * K, (unsigned)(size_t)(void*)sA[0], K);
#else
#pragma unroll
  for (int i = 0; i < 2; ++i) {
    const int c = tid + i * 256, row = c >> 2, kc = (c & 3) << 3;
    ra[i] = *(const h8v*)(A + (size_t)(m0 + row) * K + kc);
    *(h8v*)(sA[0] + row * LDH + kc) = ra[i];
  }
#endif
#pragma unroll
  for (int i = 0; i < 2; ++i) {
    const int c = tid + i * 256, kr = c >> 4, nc = (c & 15) << 3;
    rb[i] = *(const h8v*)(W + (size_t)kr * N + n0 + nc);
#pragma unroll
    for (int j = 0; j < 8; ++j) sB[0][(nc + j) * LDH + kr] = rb[i][j];
  }
#if HAS_TDM
  if (wave == 0) __builtin_amdgcn_s_wait_tensorcnt(0);
#endif
  __syncthreads();

  int p = 0;
  for (int k0 = 0; k0 < K; k0 += BK) {
    const bool have_next = (k0 + BK < K);
    if (have_next) {
      // stage next tile while this tile computes
#if HAS_TDM
      if (wave == 0)
        tdm_load_tile_a(A + (size_t)m0 * K + (k0 + BK),
                        (unsigned)(size_t)(void*)sA[p ^ 1], K);
#else
#pragma unroll
      for (int i = 0; i < 2; ++i) {
        const int c = tid + i * 256, row = c >> 2, kc = (c & 3) << 3;
        ra[i] = *(const h8v*)(A + (size_t)(m0 + row) * K + (k0 + BK) + kc);
      }
#endif
#pragma unroll
      for (int i = 0; i < 2; ++i) {
        const int c = tid + i * 256, kr = c >> 4, nc = (c & 15) << 3;
        rb[i] = *(const h8v*)(W + (size_t)(k0 + BK + kr) * N + n0 + nc);
      }
    }

    // ---- compute on buffer p ----
    v16h af[4], bf[2];
#pragma unroll
    for (int mt = 0; mt < 4; ++mt)
      af[mt] = load_a_frag(sA[p], LDH, wave_m + mt * 16, 0, lane);
#pragma unroll
    for (int nt = 0; nt < 2; ++nt)
      bf[nt] = load_b_frag(sB[p], LDH, wave_n + nt * 16, 0, lane);
#pragma unroll
    for (int mt = 0; mt < 4; ++mt)
#pragma unroll
      for (int nt = 0; nt < 2; ++nt)
        acc[mt][nt] = __builtin_amdgcn_wmma_f32_16x16x32_f16(
            false, af[mt], false, bf[nt], (short)0, acc[mt][nt], false, false);

    if (have_next) {
#pragma unroll
      for (int i = 0; i < 2; ++i) {
        const int c = tid + i * 256, kr = c >> 4, nc = (c & 15) << 3;
#pragma unroll
        for (int j = 0; j < 8; ++j) sB[p ^ 1][(nc + j) * LDH + kr] = rb[i][j];
      }
#if !HAS_TDM
#pragma unroll
      for (int i = 0; i < 2; ++i) {
        const int c = tid + i * 256, row = c >> 2, kc = (c & 3) << 3;
        *(h8v*)(sA[p ^ 1] + row * LDH + kc) = ra[i];
      }
#else
      if (wave == 0) __builtin_amdgcn_s_wait_tensorcnt(0);
#endif
    }
    __syncthreads();
    p ^= 1;
  }

  // ---- epilogue: C layout -> row j + 8*(lane/16), col lane%16 ----
#pragma unroll
  for (int mt = 0; mt < 4; ++mt)
#pragma unroll
    for (int nt = 0; nt < 2; ++nt)
#pragma unroll
      for (int j = 0; j < 8; ++j) {
        const int row = m0 + wave_m + mt * 16 + j + 8 * half;
        const int col = n0 + wave_n + nt * 16 + lr;
        float v = acc[mt][nt][j] + bias[col];
        if (resid) v += resid[(size_t)row * N + col];
        if (act) v = gelu_exact(v);
        if (Ch) Ch[(size_t)row * N + col] = (_Float16)v;
        else    Cf[(size_t)row * N + col] = v;
      }
}

// ---------------------------------------------------------------------------
// Flash attention (f16 in / f16 out): per (b,h), 64-query tile per block,
// 4 waves x 16 rows. hd = 64, scale = 1/8, online softmax.
// ---------------------------------------------------------------------------
__global__ __launch_bounds__(128) void flash_attn_wmma(
    const _Float16* __restrict__ Q, const _Float16* __restrict__ Kb,
    const _Float16* __restrict__ Vb, _Float16* __restrict__ O, int S, int H) {
  constexpr int HD = 64, LDH = 72;
  const int D = H * HD;                      // 512
  __shared__ _Float16 sQ[64 * LDH];          // [q][d]
  __shared__ _Float16 sK[64 * LDH];          // [k][d]
  __shared__ _Float16 sVT[64 * LDH];         // [d][k]
  __shared__ _Float16 sP[4 * 16 * LDH];      // per-wave P tile [16][64]

  const int tid = threadIdx.x, wave = tid >> 5, lane = tid & 31;
  const int half = lane >> 4, lr = lane & 15;
  const int q0 = blockIdx.x * 64;
  const int b = blockIdx.y / H, h = blockIdx.y % H;
  const size_t headoff = (size_t)b * S * D + (size_t)h * HD;
  const _Float16* Qp = Q + headoff;
  const _Float16* Kp = Kb + headoff;
  const _Float16* Vp = Vb + headoff;
  _Float16* Op = O + headoff;
  const float scale = 0.125f;

  // Q tile 64x64 f16: 512 chunks of 8 halves, 4 per thread.
#pragma unroll
  for (int i = 0; i < 4; ++i) {
    const int f = tid + i * 128, row = f >> 3, dc = (f & 7) << 3;
    *(h8v*)(sQ + row * LDH + dc) = *(const h8v*)(Qp + (size_t)(q0 + row) * D + dc);
  }
  __syncthreads();

  v16h aq[2];
#pragma unroll
  for (int ki = 0; ki < 2; ++ki)
    aq[ki] = load_a_frag(sQ, LDH, wave * 16, ki * 32, lane);

  v8f Oacc[4];
#pragma unroll
  for (int dt = 0; dt < 4; ++dt)
#pragma unroll
    for (int e = 0; e < 8; ++e) Oacc[dt][e] = 0.0f;
  float mrow[8], lrow[8];
#pragma unroll
  for (int j = 0; j < 8; ++j) { mrow[j] = -3.0e38f; lrow[j] = 0.0f; }

  for (int kt = 0; kt < S; kt += 64) {
#pragma unroll
    for (int i = 0; i < 4; ++i) {
      const int f = tid + i * 128, row = f >> 3, dc = (f & 7) << 3;
      *(h8v*)(sK + row * LDH + dc) = *(const h8v*)(Kp + (size_t)(kt + row) * D + dc);
      const h8v vv = *(const h8v*)(Vp + (size_t)(kt + row) * D + dc);
#pragma unroll
      for (int j = 0; j < 8; ++j) sVT[(dc + j) * LDH + row] = vv[j];
      if (kt + 64 < S) {
        __builtin_prefetch(Kp + (size_t)(kt + 64 + row) * D + dc, 0, 2);
        __builtin_prefetch(Vp + (size_t)(kt + 64 + row) * D + dc, 0, 2);
      }
    }
    __syncthreads();

    // S = (Q @ K^T) * scale
    v8f sc[4];
#pragma unroll
    for (int nt = 0; nt < 4; ++nt) {
#pragma unroll
      for (int e = 0; e < 8; ++e) sc[nt][e] = 0.0f;
#pragma unroll
      for (int ki = 0; ki < 2; ++ki) {
        v16h bk = load_b_frag(sK, LDH, nt * 16, ki * 32, lane);
        sc[nt] = __builtin_amdgcn_wmma_f32_16x16x32_f16(
            false, aq[ki], false, bk, (short)0, sc[nt], false, false);
      }
#pragma unroll
      for (int j = 0; j < 8; ++j) sc[nt][j] *= scale;
    }

    // Online softmax; row j+8*half lives in a 16-lane half -> shfl width 16.
    float mnew[8], alpha[8], rs[8];
#pragma unroll
    for (int j = 0; j < 8; ++j) {
      float t = -3.0e38f;
#pragma unroll
      for (int nt = 0; nt < 4; ++nt) t = fmaxf(t, sc[nt][j]);
#pragma unroll
      for (int off = 8; off; off >>= 1) t = fmaxf(t, __shfl_xor(t, off, 16));
      mnew[j] = fmaxf(mrow[j], t);
      alpha[j] = __expf(mrow[j] - mnew[j]);
      rs[j] = 0.0f;
    }
#pragma unroll
    for (int nt = 0; nt < 4; ++nt)
#pragma unroll
      for (int j = 0; j < 8; ++j) {
        const float pv = __expf(sc[nt][j] - mnew[j]);
        rs[j] += pv;
        sP[(wave * 16 + j + 8 * half) * LDH + nt * 16 + lr] = (_Float16)pv;
      }
#pragma unroll
    for (int j = 0; j < 8; ++j) {
#pragma unroll
      for (int off = 8; off; off >>= 1) rs[j] += __shfl_xor(rs[j], off, 16);
      lrow[j] = lrow[j] * alpha[j] + rs[j];
      mrow[j] = mnew[j];
#pragma unroll
      for (int dt = 0; dt < 4; ++dt) Oacc[dt][j] *= alpha[j];
    }

    // Wave-local P LDS turnaround before A-frag reload.
    asm volatile("s_wait_dscnt 0" ::: "memory");

    // O += P @ V
    v16h ap[2];
#pragma unroll
    for (int ki = 0; ki < 2; ++ki)
      ap[ki] = load_a_frag(sP + wave * 16 * LDH, LDH, 0, ki * 32, lane);
#pragma unroll
    for (int dt = 0; dt < 4; ++dt)
#pragma unroll
      for (int ki = 0; ki < 2; ++ki) {
        v16h bv = load_b_frag(sVT, LDH, dt * 16, ki * 32, lane);
        Oacc[dt] = __builtin_amdgcn_wmma_f32_16x16x32_f16(
            false, ap[ki], false, bv, (short)0, Oacc[dt], false, false);
      }
    __syncthreads();
  }

#pragma unroll
  for (int dt = 0; dt < 4; ++dt)
#pragma unroll
    for (int j = 0; j < 8; ++j) {
      const int row = q0 + wave * 16 + j + 8 * half;
      Op[(size_t)row * D + dt * 16 + lr] = (_Float16)(Oacc[dt][j] / lrow[j]);
    }
}

// ---------------------------------------------------------------------------
// LayerNorm over 512; writes f32 (residual path) and f16 (next-GEMM input).
// ---------------------------------------------------------------------------
__global__ __launch_bounds__(128) void ln_kernel(
    const float* __restrict__ X, const float* __restrict__ g,
    const float* __restrict__ bb, float* __restrict__ Y,
    _Float16* __restrict__ Yh) {
  constexpr int D = 512;
  const int row = blockIdx.x, tid = threadIdx.x;
  const int wave = tid >> 5, lane = tid & 31;
  const float* x = X + (size_t)row * D;
  float v[4]; float s = 0.0f;
#pragma unroll
  for (int i = 0; i < 4; ++i) { v[i] = x[tid + i * 128]; s += v[i]; }
#pragma unroll
  for (int off = 16; off; off >>= 1) s += __shfl_xor(s, off, 32);
  __shared__ float red[8];
  if (lane == 0) red[wave] = s;
  __syncthreads();
  s = red[0] + red[1] + red[2] + red[3];
  const float mean = s * (1.0f / D);
  float vs = 0.0f;
#pragma unroll
  for (int i = 0; i < 4; ++i) { const float d = v[i] - mean; vs += d * d; }
#pragma unroll
  for (int off = 16; off; off >>= 1) vs += __shfl_xor(vs, off, 32);
  __syncthreads();
  if (lane == 0) red[4 + wave] = vs;
  __syncthreads();
  vs = red[4] + red[5] + red[6] + red[7];
  const float rstd = rsqrtf(vs * (1.0f / D) + 1e-6f);
#pragma unroll
  for (int i = 0; i < 4; ++i) {
    const int c = tid + i * 128;
    const float y = (v[i] - mean) * rstd * g[c] + bb[c];
    Y[(size_t)row * D + c] = y;
    Yh[(size_t)row * D + c] = (_Float16)y;
  }
}

// ---------------------------------------------------------------------------
// Orchestration
// ---------------------------------------------------------------------------
extern "C" void kernel_launch(void* const* d_in, const int* in_sizes, int n_in,
                              void* d_out, int out_size, void* d_ws, size_t ws_size,
                              hipStream_t stream) {
  (void)in_sizes; (void)n_in; (void)out_size; (void)ws_size;
  const float* x    = (const float*)d_in[0];
  const float* t_wq = (const float*)d_in[1];  const float* t_bq = (const float*)d_in[2];
  const float* t_wk = (const float*)d_in[3];  const float* t_bk = (const float*)d_in[4];
  const float* t_wv = (const float*)d_in[5];  const float* t_bv = (const float*)d_in[6];
  const float* t_wo = (const float*)d_in[7];  const float* t_bo = (const float*)d_in[8];
  const float* s_wq = (const float*)d_in[9];  const float* s_bq = (const float*)d_in[10];
  const float* s_wk = (const float*)d_in[11]; const float* s_bk = (const float*)d_in[12];
  const float* s_wv = (const float*)d_in[13]; const float* s_bv = (const float*)d_in[14];
  const float* s_wo = (const float*)d_in[15]; const float* s_bo = (const float*)d_in[16];
  const float* ln1_g = (const float*)d_in[17]; const float* ln1_b = (const float*)d_in[18];
  const float* ln2_g = (const float*)d_in[19]; const float* ln2_b = (const float*)d_in[20];
  const float* f_w1 = (const float*)d_in[21]; const float* f_b1 = (const float*)d_in[22];
  const float* f_w2 = (const float*)d_in[23]; const float* f_b2 = (const float*)d_in[24];
  float* out = (float*)d_out;

  const int Bv = 4, Sv = 2048, Dv = 512, Hn = 8, Fv = 2048;
  const int M = Bv * Sv;                     // 8192
  const size_t seg = (size_t)M * Dv;         // 4,194,304 elems
  const size_t wsz = (size_t)Dv * Dv;        // 262,144
  const size_t fsz = (size_t)Dv * Fv;        // 1,048,576

  // f32 region
  float* bufT  = (float*)d_ws;               // pre-LN sums
  float* bufH  = bufT + seg;                 // h (residual)
  float* bufH2 = bufH + seg;                 // h2 (residual)
  // f16 region
  _Float16* fp = (_Float16*)(bufH2 + seg);
  _Float16* xh  = fp; fp += seg;             // x in f16
  _Float16* hh  = fp; fp += seg;             // h in f16
  _Float16* h2h = fp; fp += seg;             // h2 in f16
  _Float16* Qh  = fp; fp += seg;
  _Float16* Kh  = fp; fp += seg;
  _Float16* Vh  = fp; fp += seg;
  _Float16* Ah  = fp; fp += seg;             // attention output
  _Float16* Mh  = fp; fp += (size_t)M * Fv;  // FFN mid
  _Float16* wtq = fp; fp += wsz;  _Float16* wtk = fp; fp += wsz;
  _Float16* wtv = fp; fp += wsz;  _Float16* wto = fp; fp += wsz;
  _Float16* wsq = fp; fp += wsz;  _Float16* wsk = fp; fp += wsz;
  _Float16* wsv = fp; fp += wsz;  _Float16* wso = fp; fp += wsz;
  _Float16* w1h = fp; fp += fsz;  _Float16* w2h = fp; fp += fsz;

  const dim3 blk(256);
  const dim3 gD(Dv / 128, M / 128);
  const dim3 gF1(Fv / 128, M / 128);
  const dim3 fgrid(Sv / 64, Bv * Hn);
  const dim3 fblk(128);
  const dim3 lgrid(M);
  const dim3 lblk(128);

  // ---- one-shot f32 -> f16 conversions ----
  cvt_f32_f16<<<dim3(seg / 1024), blk, 0, stream>>>(x, xh, (int)seg);
  cvt_f32_f16<<<dim3(wsz / 1024), blk, 0, stream>>>(t_wq, wtq, (int)wsz);
  cvt_f32_f16<<<dim3(wsz / 1024), blk, 0, stream>>>(t_wk, wtk, (int)wsz);
  cvt_f32_f16<<<dim3(wsz / 1024), blk, 0, stream>>>(t_wv, wtv, (int)wsz);
  cvt_f32_f16<<<dim3(wsz / 1024), blk, 0, stream>>>(t_wo, wto, (int)wsz);
  cvt_f32_f16<<<dim3(wsz / 1024), blk, 0, stream>>>(s_wq, wsq, (int)wsz);
  cvt_f32_f16<<<dim3(wsz / 1024), blk, 0, stream>>>(s_wk, wsk, (int)wsz);
  cvt_f32_f16<<<dim3(wsz / 1024), blk, 0, stream>>>(s_wv, wsv, (int)wsz);
  cvt_f32_f16<<<dim3(wsz / 1024), blk, 0, stream>>>(s_wo, wso, (int)wsz);
  cvt_f32_f16<<<dim3(fsz / 1024), blk, 0, stream>>>(f_w1, w1h, (int)fsz);
  cvt_f32_f16<<<dim3(fsz / 1024), blk, 0, stream>>>(f_w2, w2h, (int)fsz);

  // ---- temporal attention ----
  gemm_bias_act<<<gD, blk, 0, stream>>>(xh, wtq, t_bq, nullptr, nullptr, Qh, M, Dv, Dv, 0);
  gemm_bias_act<<<gD, blk, 0, stream>>>(xh, wtk, t_bk, nullptr, nullptr, Kh, M, Dv, Dv, 0);
  gemm_bias_act<<<gD, blk, 0, stream>>>(xh, wtv, t_bv, nullptr, nullptr, Vh, M, Dv, Dv, 0);
  flash_attn_wmma<<<fgrid, fblk, 0, stream>>>(Qh, Kh, Vh, Ah, Sv, Hn);
  gemm_bias_act<<<gD, blk, 0, stream>>>(Ah, wto, t_bo, x, bufT, nullptr, M, Dv, Dv, 0);
  ln_kernel<<<lgrid, lblk, 0, stream>>>(bufT, ln1_g, ln1_b, bufH, hh);

  // ---- spatial attention ----
  gemm_bias_act<<<gD, blk, 0, stream>>>(hh, wsq, s_bq, nullptr, nullptr, Qh, M, Dv, Dv, 0);
  gemm_bias_act<<<gD, blk, 0, stream>>>(hh, wsk, s_bk, nullptr, nullptr, Kh, M, Dv, Dv, 0);
  gemm_bias_act<<<gD, blk, 0, stream>>>(hh, wsv, s_bv, nullptr, nullptr, Vh, M, Dv, Dv, 0);
  flash_attn_wmma<<<fgrid, fblk, 0, stream>>>(Qh, Kh, Vh, Ah, Sv, Hn);
  gemm_bias_act<<<gD, blk, 0, stream>>>(Ah, wso, s_bo, bufH, bufT, nullptr, M, Dv, Dv, 0);
  ln_kernel<<<lgrid, lblk, 0, stream>>>(bufT, ln2_g, ln2_b, bufH2, h2h);

  // ---- FFN (exact GELU) with residual into d_out ----
  gemm_bias_act<<<gF1, blk, 0, stream>>>(h2h, w1h, f_b1, nullptr, nullptr, Mh, M, Fv, Dv, 1);
  gemm_bias_act<<<gD,  blk, 0, stream>>>(Mh, w2h, f_b2, bufH2, out, nullptr, M, Dv, Fv, 0);
}
